// LSTMActorCritic_86114094284833
// MI455X (gfx1250) — matrix operand order repaired and checked by reading, not compiled, not run on hardware
//
#include <hip/hip_runtime.h>
#include <hip/hip_bf16.h>

// ---------------------------------------------------------------------------
// LSTM actor-critic forward, fully fused, CDNA5 (gfx1250) wave32 WMMA bf16.
// Grid: 128 WGs (one 16-row batch tile each). Block: 256 threads = 8 waves.
// ---------------------------------------------------------------------------

#define T_STEPS 256
#define BATCH   2048
#define OBS     64
#define HID     64
#define LSTMD   64
#define GATES   256   // 4*LSTM
#define NA      8

typedef __attribute__((ext_vector_type(16))) __bf16 v16bf;
typedef __attribute__((ext_vector_type(8)))  __bf16 v8bf;
typedef __attribute__((ext_vector_type(4)))  __bf16 v4bf;
typedef __attribute__((ext_vector_type(8)))  float  v8f;
typedef __attribute__((ext_vector_type(4)))  float  v4f;

__device__ __forceinline__ v8f wmma_bf16(v16bf a, v16bf b, v8f c) {
  // D = A(16x32 bf16) * B(32x16 bf16) + C(16x16 f32)
  return __builtin_amdgcn_wmma_f32_16x16x32_bf16(
      /*neg_a=*/false, a, /*neg_b=*/false, b,
      /*c_mod=*/(short)0, c, /*reuse_a=*/false, /*reuse_b=*/false);
}

// --- fast transcendentals: v_exp_f32 + v_rcp_f32, no IEEE div chains ------
__device__ __forceinline__ float fast_rcp(float x) {
  return __builtin_amdgcn_rcpf(x);
}
__device__ __forceinline__ float sigm(float x) {
  return fast_rcp(1.0f + __expf(-x));
}
__device__ __forceinline__ float tanh_fast(float x) {
  // tanh(x) = 2*sigmoid(2x) - 1
  return __builtin_fmaf(2.0f, fast_rcp(1.0f + __expf(-2.0f * x)), -1.0f);
}
// flax default gelu (tanh approximation)
__device__ __forceinline__ float gelu_tanh(float x) {
  float u = 0.7978845608028654f * (x + 0.044715f * x * x * x);
  return 0.5f * x * (1.0f + tanh_fast(u));
}

// --- A fragment (16x32 bf16, M x K) from row-major bf16 LDS tile -----------
// lane<16 : m=lane,    comps 0..15 = K kc*32+0..15
// lane>=16: m=lane-16, comps 0..15 = K kc*32+16..31
__device__ __forceinline__ v16bf load_a_lds(const __bf16* base, int stride,
                                            int kchunk, int lane) {
  const int m    = lane & 15;
  const int half = lane >> 4;
  const __bf16* p = base + m * stride + kchunk * 32 + 16 * half;
  v8bf lo = *(const v8bf*)(p);
  v8bf hi = *(const v8bf*)(p + 8);
  return __builtin_shufflevector(lo, hi, 0, 1, 2, 3, 4, 5, 6, 7,
                                 8, 9, 10, 11, 12, 13, 14, 15);
}

// --- B fragment (32x16 bf16, K x N) from row-major (K x ncols) f32 weights -
// lane l: n = ntile*16 + (l&15); comps c=0..15 -> K = kchunk*32 + 16*(l>>4) + c
__device__ __forceinline__ v16bf load_b_f32(const float* W, int ncols,
                                            int ntile, int kchunk, int lane) {
  const int n     = ntile * 16 + (lane & 15);
  const int kbase = kchunk * 32 + ((lane >> 4) << 4);
  v16bf b;
#pragma unroll
  for (int c = 0; c < 16; ++c)
    b[c] = (__bf16)W[(size_t)(kbase + c) * ncols + n];
  return b;
}

// combined head weights: cols 0..7 = actor_w (64x8), col 8 = value_w (64x1)
__device__ __forceinline__ v16bf load_b_head(const float* aw, const float* vw,
                                             int kchunk, int lane) {
  const int n     = lane & 15;
  const int kbase = kchunk * 32 + ((lane >> 4) << 4);
  v16bf b;
#pragma unroll
  for (int c = 0; c < 16; ++c) {
    float v = 0.0f;
    if (n < 8)       v = aw[(size_t)(kbase + c) * NA + n];
    else if (n == 8) v = vw[kbase + c];
    b[c] = (__bf16)v;
  }
  return b;
}

__device__ __forceinline__ v8f splat8(float v) {
  v8f r;
#pragma unroll
  for (int j = 0; j < 8; ++j) r[j] = v;
  return r;
}

__global__ __launch_bounds__(256)
void lstm_ac_fused_kernel(const float* __restrict__ x,
                          const float* __restrict__ enc_w1,
                          const float* __restrict__ enc_b1,
                          const float* __restrict__ enc_w2,
                          const float* __restrict__ enc_b2,
                          const float* __restrict__ wi,
                          const float* __restrict__ wh,
                          const float* __restrict__ bh,
                          const float* __restrict__ actor_w,
                          const float* __restrict__ actor_b,
                          const float* __restrict__ value_w,
                          const float* __restrict__ value_b,
                          const float* __restrict__ log_std,
                          float* __restrict__ out) {
  __shared__ __align__(16) __bf16 xs[16 * OBS];     // current x tile, bf16
  __shared__ __align__(16) __bf16 feat1[16 * HID];  // gelu(x@w1+b1)
  __shared__ __align__(16) __bf16 feat2[16 * HID];  // encoder output
  __shared__ __align__(16) __bf16 hbuf[16 * LSTMD]; // hidden state, bf16
  __shared__ __align__(16) float  gbuf[16 * GATES]; // fused gates, f32

  const int tid  = threadIdx.x;
  const int lane = tid & 31;
  const int wave = tid >> 5;
  const int nlo  = lane & 15;
  const int half = lane >> 4;
  const int b0   = blockIdx.x * 16;  // batch tile origin

  float* d_means  = out;                                   // (T,B,A)
  float* d_values = out + (size_t)T_STEPS * BATCH * NA;    // (T,B)
  float* d_std    = d_values + (size_t)T_STEPS * BATCH;    // (A)

  if (blockIdx.x == 0 && tid < NA) d_std[tid] = __expf(log_std[tid]);

  // ---- preload weight B-fragments into registers (once) ----
  v16bf w1f[2], w2f[2], wif[2][2], whf[2][2], headf[2];
  float b1s = 0.f, b2s = 0.f, bhs[2], headb = 0.f;

  if (wave < 4) {
#pragma unroll
    for (int kc = 0; kc < 2; ++kc) {
      w1f[kc] = load_b_f32(enc_w1, HID, wave, kc, lane);
      w2f[kc] = load_b_f32(enc_w2, HID, wave, kc, lane);
    }
    b1s = enc_b1[wave * 16 + nlo];
    b2s = enc_b2[wave * 16 + nlo];
  }
#pragma unroll
  for (int q = 0; q < 2; ++q) {
    const int nt = 2 * wave + q;
#pragma unroll
    for (int kc = 0; kc < 2; ++kc) {
      wif[q][kc] = load_b_f32(wi, GATES, nt, kc, lane);
      whf[q][kc] = load_b_f32(wh, GATES, nt, kc, lane);
    }
    bhs[q] = bh[nt * 16 + nlo];
  }
  if (wave == 0) {
#pragma unroll
    for (int kc = 0; kc < 2; ++kc)
      headf[kc] = load_b_head(actor_w, value_w, kc, lane);
    headb = (nlo < 8) ? actor_b[nlo] : ((nlo == 8) ? value_b[0] : 0.0f);
  }

  // ---- zero initial hidden/cell state ----
  {
    const int e = tid * 4;
#pragma unroll
    for (int k = 0; k < 4; ++k) hbuf[e + k] = (__bf16)0.0f;
  }
  float c_reg[4] = {0.f, 0.f, 0.f, 0.f};
  __syncthreads();

  // elementwise ownership: 4 CONSECUTIVE columns per thread -> vector LDS ops
  const int em = tid >> 4;         // row 0..15
  const int ej = (tid & 15) * 4;   // first of 4 consecutive columns

  for (int t = 0; t < T_STEPS; ++t) {
    // ---- stage 1: stream x tile -> LDS bf16 (tile is contiguous in memory)
    {
      const float* xt = x + ((size_t)t * BATCH + b0) * OBS;
      const int e = tid * 4;
      v4f v = *(const v4f*)(xt + e);
      v4bf xb;
#pragma unroll
      for (int k = 0; k < 4; ++k) xb[k] = (__bf16)v[k];
      *(v4bf*)(xs + e) = xb;
      if (t + 1 < T_STEPS)
        __builtin_prefetch(xt + (size_t)BATCH * OBS + e, 0, 0);
    }
    __syncthreads();

    // ---- stage 2: encoder layer 1 (waves 0..3, one N-tile each) ----
    if (wave < 4) {
      v8f acc = splat8(b1s);
#pragma unroll
      for (int kc = 0; kc < 2; ++kc)
        acc = wmma_bf16(load_a_lds(xs, OBS, kc, lane), w1f[kc], acc);
#pragma unroll
      for (int j = 0; j < 8; ++j)
        feat1[(j + 8 * half) * HID + wave * 16 + nlo] =
            (__bf16)gelu_tanh(acc[j]);
    }
    __syncthreads();

    // ---- stage 3: encoder layer 2 ----
    if (wave < 4) {
      v8f acc = splat8(b2s);
#pragma unroll
      for (int kc = 0; kc < 2; ++kc)
        acc = wmma_bf16(load_a_lds(feat1, HID, kc, lane), w2f[kc], acc);
#pragma unroll
      for (int j = 0; j < 8; ++j)
        feat2[(j + 8 * half) * HID + wave * 16 + nlo] =
            (__bf16)gelu_tanh(acc[j]);
    }
    __syncthreads();

    // ---- stage 4: gates = feats@wi + bh + h@wh   (all 8 waves, 2 tiles) --
    {
      v16bf af[2], ah[2];
#pragma unroll
      for (int kc = 0; kc < 2; ++kc) {
        af[kc] = load_a_lds(feat2, HID, kc, lane);
        ah[kc] = load_a_lds(hbuf, LSTMD, kc, lane);
      }
#pragma unroll
      for (int q = 0; q < 2; ++q) {
        v8f acc = splat8(bhs[q]);
#pragma unroll
        for (int kc = 0; kc < 2; ++kc) acc = wmma_bf16(af[kc], wif[q][kc], acc);
#pragma unroll
        for (int kc = 0; kc < 2; ++kc) acc = wmma_bf16(ah[kc], whf[q][kc], acc);
        const int nt = 2 * wave + q;
#pragma unroll
        for (int j = 0; j < 8; ++j)
          gbuf[(j + 8 * half) * GATES + nt * 16 + nlo] = acc[j];
      }
    }
    __syncthreads();

    // ---- stage 5: elementwise LSTM cell: 4 ds_load_b128 + 1 ds_store_b64 -
    {
      const float* gr = gbuf + em * GATES + ej;
      const v4f ig4 = *(const v4f*)(gr);
      const v4f fg4 = *(const v4f*)(gr + 64);
      const v4f gg4 = *(const v4f*)(gr + 128);
      const v4f og4 = *(const v4f*)(gr + 192);
      v4bf hb;
#pragma unroll
      for (int k = 0; k < 4; ++k) {
        const float c =
            sigm(fg4[k]) * c_reg[k] + sigm(ig4[k]) * tanh_fast(gg4[k]);
        c_reg[k] = c;
        hb[k] = (__bf16)(sigm(og4[k]) * tanh_fast(c));
      }
      *(v4bf*)(hbuf + em * LSTMD + ej) = hb;
    }
    __syncthreads();

    // ---- stage 6: actor/value head (wave 0): h @ [actor_w | value_w | 0] -
    if (wave == 0) {
      v8f acc = splat8(headb);
#pragma unroll
      for (int kc = 0; kc < 2; ++kc)
        acc = wmma_bf16(load_a_lds(hbuf, LSTMD, kc, lane), headf[kc], acc);
#pragma unroll
      for (int j = 0; j < 8; ++j) {
        const int m = j + 8 * half;
        const size_t row = (size_t)t * BATCH + b0 + m;
        if (nlo < 8)        d_means[row * NA + nlo] = acc[j];
        else if (nlo == 8)  d_values[row] = acc[j];
      }
    }
    // no barrier needed: stage-1 barrier of next iteration orders hbuf reuse
  }
}

extern "C" void kernel_launch(void* const* d_in, const int* in_sizes, int n_in,
                              void* d_out, int out_size, void* d_ws, size_t ws_size,
                              hipStream_t stream) {
  (void)in_sizes; (void)n_in; (void)d_ws; (void)ws_size; (void)out_size;
  const float* x       = (const float*)d_in[0];
  const float* enc_w1  = (const float*)d_in[1];
  const float* enc_b1  = (const float*)d_in[2];
  const float* enc_w2  = (const float*)d_in[3];
  const float* enc_b2  = (const float*)d_in[4];
  const float* wi      = (const float*)d_in[5];
  const float* wh      = (const float*)d_in[6];
  const float* bh      = (const float*)d_in[7];
  const float* actor_w = (const float*)d_in[8];
  const float* actor_b = (const float*)d_in[9];
  const float* value_w = (const float*)d_in[10];
  const float* value_b = (const float*)d_in[11];
  const float* log_std = (const float*)d_in[12];
  float* out = (float*)d_out;

  dim3 grid(BATCH / 16);   // 128 workgroups, one 16-row batch tile each
  dim3 block(256);         // 8 wave32 waves
  lstm_ac_fused_kernel<<<grid, block, 0, stream>>>(
      x, enc_w1, enc_b1, enc_w2, enc_b2, wi, wh, bh,
      actor_w, actor_b, value_w, value_b, log_std, out);
}
